// MultilevelEncoder_36352603193532
// MI455X (gfx1250) — matrix operand order repaired and checked by reading, not compiled, not run on hardware
//
#include <hip/hip_runtime.h>
#include <hip/hip_bf16.h>

#define BB 64
#define TT 128
#define FF 3072
#define DD 1024
#define CV 512
#define CN 1024
#define NV 10
#define NN 20
#define LDA 40   // LDS row pitch in bf16 elems (32 data + 8 pad, 80B keeps 16B alignment)

typedef __attribute__((ext_vector_type(16))) __bf16 v16bf;
typedef __attribute__((ext_vector_type(8)))  __bf16 v8bf;
typedef __attribute__((ext_vector_type(8)))  float  v8f;

__device__ __forceinline__ v16bf frag_cat(const __bf16* p0, const __bf16* p1) {
  v8bf lo = *(const v8bf*)p0;
  v8bf hi = *(const v8bf*)p1;
  v16bf t;
#pragma unroll
  for (int i = 0; i < 8; ++i) { t[i] = lo[i]; t[i + 8] = hi[i]; }
  return t;
}

__device__ __forceinline__ float bn_sig(float x, float bias, float g, float beta,
                                        float m, float v) {
  float inv = rsqrtf(v + 1e-5f);
  float y = (x + bias - m) * (g * inv) + beta;
  return 1.f / (1.f + expf(-y));
}

// ---------------------------------------------------------------------------
// WMMA bf16 GEMM: Out[M,N] = A[M,K] * W (+epilogue).
// transB==0: W is [K,N] row-major (embed).  transB==1: W is [N,K] (noun conv).
// EPI 0: Out = acc + p0[n].  EPI 1: Out = sigmoid(BN(acc + p0[n])).
// Block: 256 thr = 8 waves, tile 128(M) x 128(N); wave tile 64x32 (8 accums).
// LDS double-buffered: one barrier per K-step, next tile's loads overlap WMMA.
// ---------------------------------------------------------------------------
template<int EPI>
__global__ void gemm_kernel(const float* __restrict__ A, const float* __restrict__ W,
                            const float* __restrict__ p0, const float* __restrict__ p1,
                            const float* __restrict__ p2, const float* __restrict__ p3,
                            const float* __restrict__ p4,
                            float* __restrict__ Out, int Mtot, int N, int K, int transB) {
  const int tid  = threadIdx.x;
  const int lane = tid & 31;
  const int wid  = tid >> 5;
  const int waveM = wid & 1;        // 2 waves along M (2 x 64 = 128)
  const int waveN = wid >> 1;       // 4 waves along N (4 x 32 = 128)
  const int lm = lane & 15;
  const int lh = lane >> 4;
  const int m0 = blockIdx.y * 128;
  const int n0 = blockIdx.x * 128;

  __shared__ __bf16 As[2][128 * LDA];
  __shared__ __bf16 Bs[2][128 * LDA];

  v8f acc[8];
#pragma unroll
  for (int j = 0; j < 8; ++j) acc[j] = 0.0f;

  const int sr  = tid >> 1;               // staging row (A and B)
  const int sc0 = (tid & 1) * 16;         // 16 staged cols

  auto stage = [&](int k0, int buf) {
    {
      const float* src = A + (size_t)(m0 + sr) * K + k0 + sc0;
      __builtin_prefetch(src + 32, 0, 1);          // next K tile -> global_prefetch_b8
#pragma unroll
      for (int i = 0; i < 16; ++i) As[buf][sr * LDA + sc0 + i] = (__bf16)src[i];
    }
    if (transB) {
      const float* src = W + (size_t)(n0 + sr) * K + k0 + sc0;
      __builtin_prefetch(src + 32, 0, 1);
#pragma unroll
      for (int i = 0; i < 16; ++i) Bs[buf][sr * LDA + sc0 + i] = (__bf16)src[i];
    } else {
#pragma unroll
      for (int i = 0; i < 16; ++i)
        Bs[buf][sr * LDA + sc0 + i] = (__bf16)W[(size_t)(k0 + sc0 + i) * N + n0 + sr];
    }
  };

  stage(0, 0);
  int p = 0;
  for (int k0 = 0; k0 < K; k0 += 32) {
    __syncthreads();                       // buf p staged; buf 1-p free for writes
    if (k0 + 32 < K) stage(k0 + 32, 1 - p);

    const __bf16* bp0 = &Bs[p][(waveN * 32 + lm) * LDA + lh * 16];
    const __bf16* bp1 = bp0 + 16 * LDA;
    v16bf bf0 = frag_cat(bp0, bp0 + 8);
    v16bf bf1 = frag_cat(bp1, bp1 + 8);
#pragma unroll
    for (int j = 0; j < 4; ++j) {
      const __bf16* ap = &As[p][(waveM * 64 + j * 16 + lm) * LDA + lh * 16];
      v16bf af = frag_cat(ap, ap + 8);
      acc[j * 2 + 0] = __builtin_amdgcn_wmma_f32_16x16x32_bf16(
          false, af, false, bf0, (short)0, acc[j * 2 + 0], false, false);
      acc[j * 2 + 1] = __builtin_amdgcn_wmma_f32_16x16x32_bf16(
          false, af, false, bf1, (short)0, acc[j * 2 + 1], false, false);
    }
    p ^= 1;
  }

#pragma unroll
  for (int s = 0; s < 2; ++s) {
    const int nc = n0 + waveN * 32 + s * 16 + lm;
    const float bias = p0[nc];
    float g = 0.f, beta = 0.f, mm = 0.f, vv = 0.f;
    if (EPI == 1) { g = p1[nc]; beta = p2[nc]; mm = p3[nc]; vv = p4[nc]; }
#pragma unroll
    for (int j = 0; j < 4; ++j) {
      const int mbase = m0 + waveM * 64 + j * 16 + 8 * lh;
#pragma unroll
      for (int v = 0; v < 8; ++v) {
        float x = acc[j * 2 + s][v];
        float y = (EPI == 0) ? (x + bias) : bn_sig(x, bias, g, beta, mm, vv);
        Out[(size_t)(mbase + v) * N + nc] = y;
      }
    }
  }
}

// ---------------------------------------------------------------------------
// Verb conv1d k=5 pad=2 as WMMA GEMM per batch: M=T(128) x N=128 tile of CV,
// K = 5 taps x D (flattened step loop, LDS double-buffered).
// BN + sigmoid epilogue -> logits_verb[b,t,c].
// ---------------------------------------------------------------------------
__global__ void conv_verb_kernel(const float* __restrict__ e1, const float* __restrict__ Wv,
                                 const float* __restrict__ bv, const float* __restrict__ gv,
                                 const float* __restrict__ betav, const float* __restrict__ mv,
                                 const float* __restrict__ vv, float* __restrict__ logits) {
  const int tid  = threadIdx.x;
  const int lane = tid & 31;
  const int wid  = tid >> 5;
  const int waveM = wid & 1;
  const int waveN = wid >> 1;
  const int lm = lane & 15;
  const int lh = lane >> 4;
  const int n0 = blockIdx.x * 128;
  const int b  = blockIdx.y;

  __shared__ __bf16 As[2][128 * LDA];
  __shared__ __bf16 Bs[2][128 * LDA];

  v8f acc[8];
#pragma unroll
  for (int j = 0; j < 8; ++j) acc[j] = 0.0f;

  const int sr  = tid >> 1;
  const int sc0 = (tid & 1) * 16;
  const int NSTEP = 5 * (DD / 32);

  auto stage = [&](int step, int buf) {
    const int kk = step / (DD / 32);
    const int d0 = (step % (DD / 32)) * 32;
    const int st = sr + kk - 2;                    // shifted time row, zero-padded
    if (st >= 0 && st < TT) {
      const float* src = e1 + ((size_t)b * TT + st) * DD + d0 + sc0;
#pragma unroll
      for (int i = 0; i < 16; ++i) As[buf][sr * LDA + sc0 + i] = (__bf16)src[i];
    } else {
#pragma unroll
      for (int i = 0; i < 16; ++i) As[buf][sr * LDA + sc0 + i] = (__bf16)0.0f;
    }
#pragma unroll
    for (int i = 0; i < 16; ++i) {
      int dd = d0 + sc0 + i;
      Bs[buf][sr * LDA + sc0 + i] = (__bf16)Wv[((size_t)(n0 + sr) * DD + dd) * 5 + kk];
    }
  };

  stage(0, 0);
  int p = 0;
  for (int step = 0; step < NSTEP; ++step) {
    __syncthreads();
    if (step + 1 < NSTEP) stage(step + 1, 1 - p);

    const __bf16* bp0 = &Bs[p][(waveN * 32 + lm) * LDA + lh * 16];
    const __bf16* bp1 = bp0 + 16 * LDA;
    v16bf bf0 = frag_cat(bp0, bp0 + 8);
    v16bf bf1 = frag_cat(bp1, bp1 + 8);
#pragma unroll
    for (int j = 0; j < 4; ++j) {
      const __bf16* ap = &As[p][(waveM * 64 + j * 16 + lm) * LDA + lh * 16];
      v16bf af = frag_cat(ap, ap + 8);
      acc[j * 2 + 0] = __builtin_amdgcn_wmma_f32_16x16x32_bf16(
          false, af, false, bf0, (short)0, acc[j * 2 + 0], false, false);
      acc[j * 2 + 1] = __builtin_amdgcn_wmma_f32_16x16x32_bf16(
          false, af, false, bf1, (short)0, acc[j * 2 + 1], false, false);
    }
    p ^= 1;
  }

#pragma unroll
  for (int s = 0; s < 2; ++s) {
    const int nc = n0 + waveN * 32 + s * 16 + lm;
    const float bias = bv[nc], g = gv[nc], be = betav[nc], mm = mv[nc], vr = vv[nc];
#pragma unroll
    for (int j = 0; j < 4; ++j) {
      const int mbase = waveM * 64 + j * 16 + 8 * lh;
#pragma unroll
      for (int v = 0; v < 8; ++v) {
        const int t = mbase + v;
        logits[((size_t)b * TT + t) * CV + nc] = bn_sig(acc[j * 2 + s][v], bias, g, be, mm, vr);
      }
    }
  }
}

// --------------------------- attention pooling ------------------------------
__global__ void attn_kernel(const float* __restrict__ e0, const float* __restrict__ wv,
                            const float* __restrict__ battn, const int* __restrict__ lens,
                            float* __restrict__ sent) {
  const int b = blockIdx.x, tid = threadIdx.x;   // 128 threads
  __shared__ float sc[TT];
  __shared__ float red[TT];
  const float* row = e0 + ((size_t)b * TT + tid) * DD;
  float s = 0.f;
  for (int d = 0; d < DD; d += 4)
    s += row[d] * wv[d] + row[d+1] * wv[d+1] + row[d+2] * wv[d+2] + row[d+3] * wv[d+3];
  s += battn[0];
  const int len = lens[b];
  if (tid >= len) s = -1e18f;
  red[tid] = s; __syncthreads();
  for (int st = 64; st; st >>= 1) { if (tid < st) red[tid] = fmaxf(red[tid], red[tid + st]); __syncthreads(); }
  const float mx = red[0]; __syncthreads();
  const float ex = expf(s - mx);
  sc[tid] = ex; red[tid] = ex; __syncthreads();
  for (int st = 64; st; st >>= 1) { if (tid < st) red[tid] += red[tid + st]; __syncthreads(); }
  const float inv = 1.f / red[0]; __syncthreads();
  for (int i = 0; i < 8; ++i) {
    const int d = tid + i * 128;
    float acc = 0.f;
    for (int t = 0; t < TT; ++t) acc += sc[t] * e0[((size_t)b * TT + t) * DD + d];
    sent[(size_t)b * DD + d] = acc * inv;
  }
}

// --------------------------- SE scale factors -------------------------------
__global__ void se_kernel(const float* __restrict__ e, const float* __restrict__ w1,
                          const float* __restrict__ w2, float* __restrict__ z) {
  const int b = blockIdx.x, tid = threadIdx.x;   // 256 threads
  __shared__ float y[DD];
  __shared__ float h[DD / 16];
  for (int i = 0; i < 4; ++i) {
    const int d = tid * 4 + i;
    float s = 0.f;
    for (int t = 0; t < TT; ++t) s += e[((size_t)b * TT + t) * DD + d];
    y[d] = s * (1.f / TT);
  }
  __syncthreads();
  if (tid < DD / 16) {
    float s = 0.f;
    for (int d = 0; d < DD; ++d) s += y[d] * w1[(size_t)d * (DD / 16) + tid];
    h[tid] = fmaxf(s, 0.f);
  }
  __syncthreads();
  for (int i = 0; i < 4; ++i) {
    const int d = tid * 4 + i;
    float s = 0.f;
    for (int j = 0; j < DD / 16; ++j) s += h[j] * w2[(size_t)j * DD + d];
    z[(size_t)b * DD + d] = 1.f / (1.f + expf(-s));
  }
}

// --------------------------- MIL top-k mean (one wave per (b,c)) ------------
__global__ void mil_kernel(const float* __restrict__ logits, const int* __restrict__ lens,
                           float* __restrict__ inst, int C) {
  const int gw = blockIdx.x * 8 + (threadIdx.x >> 5);
  const int lane = threadIdx.x & 31;
  const int b = gw / C, c = gw % C;
  const int len = lens[b];
  const int kk = (len + 7) >> 3;                 // 8..16
  float v[4];
#pragma unroll
  for (int i = 0; i < 4; ++i) {
    const int t = lane + i * 32;
    v[i] = (t < len) ? logits[((size_t)b * TT + t) * C + c] : -1e9f;
  }
  float sum = 0.f;
  for (int it = 0; it < kk; ++it) {
    float m = v[0]; int mi = 0;
#pragma unroll
    for (int i = 1; i < 4; ++i) if (v[i] > m) { m = v[i]; mi = i; }
    float bm = m; int bl = lane;
    for (int off = 16; off; off >>= 1) {
      float om = __shfl_xor(bm, off, 32);
      int   ol = __shfl_xor(bl, off, 32);
      if (om > bm || (om == bm && ol < bl)) { bm = om; bl = ol; }
    }
    sum += bm;
    if (lane == bl) v[mi] = -1e30f;
  }
  if (lane == 0) inst[gw] = sum / (float)kk;
}

// --------------------------- block top-K ------------------------------------
__global__ void topk_kernel(const float* __restrict__ inst, float* __restrict__ of,
                            int* __restrict__ oi, int C, int K) {
  const int b = blockIdx.x, tid = threadIdx.x;   // 256 threads
  __shared__ float vals[1024];
  __shared__ float sv[256];
  __shared__ int   si[256];
  for (int c = tid; c < C; c += 256) vals[c] = inst[(size_t)b * C + c];
  __syncthreads();
  for (int it = 0; it < K; ++it) {
    float m = -1e30f; int mi = 0;
    for (int c = tid; c < C; c += 256) if (vals[c] > m) { m = vals[c]; mi = c; }
    sv[tid] = m; si[tid] = mi; __syncthreads();
    for (int s = 128; s; s >>= 1) {
      if (tid < s && (sv[tid + s] > sv[tid] ||
                      (sv[tid + s] == sv[tid] && si[tid + s] < si[tid]))) {
        sv[tid] = sv[tid + s]; si[tid] = si[tid + s];
      }
      __syncthreads();
    }
    if (tid == 0) { of[b * K + it] = (float)si[0]; oi[b * K + it] = si[0]; vals[si[0]] = -1e30f; }
    __syncthreads();
  }
}

// --------------------------- verb argmax indices + cond ---------------------
__global__ void verb_ind_kernel(const float* __restrict__ lv, const int* __restrict__ topv,
                                const int* __restrict__ lens, int* __restrict__ indv,
                                int* __restrict__ condv) {
  const int b = blockIdx.x, tid = threadIdx.x;   // 128 threads
  __shared__ float sv[TT];
  __shared__ int   si[TT];
  __shared__ int   inds[NV];
  const int len = lens[b];
  for (int j = 0; j < NV; ++j) {
    const int c = topv[b * NV + j];
    sv[tid] = (tid < len) ? lv[((size_t)b * TT + tid) * CV + c] : -1e9f;
    si[tid] = tid;
    __syncthreads();
    for (int s = 64; s; s >>= 1) {
      if (tid < s && sv[tid + s] > sv[tid]) { sv[tid] = sv[tid + s]; si[tid] = si[tid + s]; }
      __syncthreads();
    }
    if (tid == 0) inds[j] = si[0];
    __syncthreads();
  }
  if (tid == 0) {
    int mx = inds[0], mn = inds[0];
    for (int j = 1; j < NV; ++j) { mx = max(mx, inds[j]); mn = min(mn, inds[j]); }
    condv[b] = ((mx + 2) < len) && (mn > 1);
    for (int j = 0; j < NV; ++j) indv[b * NV + j] = inds[j];
  }
}

__global__ void verb_gather_kernel(const float* __restrict__ e1, const float* __restrict__ z1,
                                   const int* __restrict__ indv, const int* __restrict__ condv,
                                   float* __restrict__ out) {
  const int bj = blockIdx.x;
  const int b  = bj / NV;
  const int t0 = indv[bj];
  const int cd = condv[b];
  for (int i = 0; i < 4; ++i) {
    const int d = threadIdx.x * 4 + i;
    const float zz = z1[(size_t)b * DD + d];
    float val;
    if (cd) {
      float s = 0.f;
      for (int o = -2; o <= 2; ++o) {
        int tt = t0 + o; tt = tt < 0 ? 0 : (tt > TT - 1 ? TT - 1 : tt);
        s += e1[((size_t)b * TT + tt) * DD + d];
      }
      val = s * 0.2f;
    } else {
      val = e1[((size_t)b * TT + t0) * DD + d];
    }
    out[(size_t)bj * DD + d] = val * zz;
  }
}

// --------------------------- noun top-3 indices + gather --------------------
__global__ void noun_ind_kernel(const float* __restrict__ ln, const int* __restrict__ topn,
                                const int* __restrict__ lens, int* __restrict__ ind3) {
  const int b = blockIdx.x, tid = threadIdx.x;   // 128 threads
  __shared__ float val[TT];
  __shared__ float sv[TT];
  __shared__ int   si[TT];
  const int len = lens[b];
  for (int j = 0; j < NN; ++j) {
    const int c = topn[b * NN + j];
    val[tid] = (tid < len) ? ln[((size_t)b * TT + tid) * CN + c] : -1e9f;
    __syncthreads();
    for (int it = 0; it < 3; ++it) {
      sv[tid] = val[tid]; si[tid] = tid;
      __syncthreads();
      for (int s = 64; s; s >>= 1) {
        if (tid < s && sv[tid + s] > sv[tid]) { sv[tid] = sv[tid + s]; si[tid] = si[tid + s]; }
        __syncthreads();
      }
      if (tid == 0) { ind3[(b * NN + j) * 3 + it] = si[0]; val[si[0]] = -1e30f; }
      __syncthreads();
    }
  }
}

__global__ void noun_gather_kernel(const float* __restrict__ e2, const float* __restrict__ z2,
                                   const int* __restrict__ ind3, const int* __restrict__ lens,
                                   float* __restrict__ out) {
  const int bj = blockIdx.x;
  const int b  = bj / NN;
  const int i0 = ind3[bj * 3 + 0], i1 = ind3[bj * 3 + 1], i2 = ind3[bj * 3 + 2];
  const int len = lens[b];
  for (int i = 0; i < 4; ++i) {
    const int d = threadIdx.x * 4 + i;
    const float zz = z2[(size_t)b * DD + d];
    float val;
    if (len > 2) {
      val = (e2[((size_t)b * TT + i0) * DD + d] + e2[((size_t)b * TT + i1) * DD + d] +
             e2[((size_t)b * TT + i2) * DD + d]) * (1.f / 3.f);
    } else {
      val = e2[((size_t)b * TT + i0) * DD + d];
    }
    out[(size_t)bj * DD + d] = val * zz;
  }
}

// ===========================================================================
extern "C" void kernel_launch(void* const* d_in, const int* in_sizes, int n_in,
                              void* d_out, int out_size, void* d_ws, size_t ws_size,
                              hipStream_t stream) {
  (void)in_sizes; (void)n_in; (void)out_size; (void)ws_size;

  const float* inputs = (const float*)d_in[0];
  const int*   lens   = (const int*)d_in[1];
  const float* Wemb   = (const float*)d_in[2];
  const float* bemb   = (const float*)d_in[3];
  const float* wattn  = (const float*)d_in[4];
  const float* battn  = (const float*)d_in[5];
  const float* Wv     = (const float*)d_in[6];
  const float* bv     = (const float*)d_in[7];
  const float* gv     = (const float*)d_in[8];
  const float* betav  = (const float*)d_in[9];
  const float* mv     = (const float*)d_in[10];
  const float* vv     = (const float*)d_in[11];
  const float* Wn     = (const float*)d_in[12];
  const float* bn_    = (const float*)d_in[13];
  const float* gn     = (const float*)d_in[14];
  const float* betan  = (const float*)d_in[15];
  const float* mn_    = (const float*)d_in[16];
  const float* vn_    = (const float*)d_in[17];
  const float* se1w1  = (const float*)d_in[18];
  const float* se1w2  = (const float*)d_in[19];
  const float* se2w1  = (const float*)d_in[20];
  const float* se2w2  = (const float*)d_in[21];

  // ---- d_out layout (tuple order, flat fp32) ----
  float* out   = (float*)d_out;
  float* o_sent = out;                        // B*D
  float* o_ev   = o_sent + (size_t)BB * DD;   // B*NV*D
  float* o_en   = o_ev + (size_t)BB * NV * DD;// B*NN*D
  float* o_e1   = o_en + (size_t)BB * NN * DD;// B*T*D
  float* o_e2   = o_e1 + (size_t)BB * TT * DD;// B*T*D
  float* o_iv   = o_e2 + (size_t)BB * TT * DD;// B*CV
  float* o_in   = o_iv + (size_t)BB * CV;     // B*CN
  float* o_tv   = o_in + (size_t)BB * CN;     // B*NV
  float* o_tn   = o_tv + (size_t)BB * NV;     // B*NN

  // ---- workspace layout ----
  float* e0       = (float*)d_ws;                       // B*T*D
  float* logits_v = e0 + (size_t)BB * TT * DD;          // B*T*CV
  float* logits_n = logits_v + (size_t)BB * TT * CV;    // B*T*CN
  float* z1       = logits_n + (size_t)BB * TT * CN;    // B*D
  float* z2       = z1 + (size_t)BB * DD;               // B*D
  int*   topv_i   = (int*)(z2 + (size_t)BB * DD);       // B*NV
  int*   topn_i   = topv_i + BB * NV;                   // B*NN
  int*   indv     = topn_i + BB * NN;                   // B*NV
  int*   condv    = indv + BB * NV;                     // B
  int*   ind3     = condv + BB;                         // B*NN*3

  const int Mtot = BB * TT;

  // 1) embedding GEMMs (bf16 WMMA): e0 -> ws, e1/e2 -> outputs
  {
    dim3 grid(DD / 128, Mtot / 128);
    float* outs[3] = { e0, o_e1, o_e2 };
    for (int lvl = 0; lvl < 3; ++lvl) {
      gemm_kernel<0><<<grid, 256, 0, stream>>>(
          inputs, Wemb + (size_t)lvl * FF * DD,
          bemb + (size_t)lvl * DD, nullptr, nullptr, nullptr, nullptr,
          outs[lvl], Mtot, DD, FF, 0);
    }
  }

  // 2) attention pooling -> sent_embeds
  attn_kernel<<<BB, 128, 0, stream>>>(e0, wattn, battn, lens, o_sent);

  // 3) SE scale factors z1 (from e1) and z2 (from e2)
  se_kernel<<<BB, 256, 0, stream>>>(o_e1, se1w1, se1w2, z1);
  se_kernel<<<BB, 256, 0, stream>>>(o_e2, se2w1, se2w2, z2);

  // 4) verb conv (k=5) via WMMA + BN + sigmoid -> logits_verb (ws)
  conv_verb_kernel<<<dim3(CV / 128, BB), 256, 0, stream>>>(
      o_e1, Wv, bv, gv, betav, mv, vv, logits_v);

  // 5) noun conv (k=1) as WMMA GEMM + BN + sigmoid -> logits_noun (ws)
  gemm_kernel<1><<<dim3(CN / 128, Mtot / 128), 256, 0, stream>>>(
      o_e2, Wn, bn_, gn, betan, mn_, vn_, logits_n, Mtot, CN, DD, 1);

  // 6) MIL top-k mean pooling -> inst_verb / inst_noun
  mil_kernel<<<(BB * CV) / 8, 256, 0, stream>>>(logits_v, lens, o_iv, CV);
  mil_kernel<<<(BB * CN) / 8, 256, 0, stream>>>(logits_n, lens, o_in, CN);

  // 7) top-K concepts
  topk_kernel<<<BB, 256, 0, stream>>>(o_iv, o_tv, topv_i, CV, NV);
  topk_kernel<<<BB, 256, 0, stream>>>(o_in, o_tn, topn_i, CN, NN);

  // 8) verb embeds: argmax frames + 5-frame mean (SE scale applied at gather)
  verb_ind_kernel<<<BB, 128, 0, stream>>>(logits_v, topv_i, lens, indv, condv);
  verb_gather_kernel<<<BB * NV, 256, 0, stream>>>(o_e1, z1, indv, condv, o_ev);

  // 9) noun embeds: top-3 frame mean (SE scale applied at gather)
  noun_ind_kernel<<<BB, 128, 0, stream>>>(logits_n, topn_i, lens, ind3);
  noun_gather_kernel<<<BB * NN, 256, 0, stream>>>(o_e2, z2, ind3, lens, o_en);
}
